// RouterBigger_1984274891210
// MI455X (gfx1250) — matrix operand order repaired and checked by reading, not compiled, not run on hardware
//
#include <hip/hip_runtime.h>

constexpr int kHidden    = 4096;
constexpr int kExperts   = 64;
constexpr int kTopK      = 8;
constexpr int kCols      = 256;   // N_EXPERTS * BIGGER
constexpr int kTokens    = 8192;
constexpr int kTokPerBlk = 32;
constexpr int kThreads   = 512;   // 16 waves of 32

typedef __attribute__((ext_vector_type(16))) __bf16          v16bf;
typedef __attribute__((ext_vector_type(8)))  unsigned int    v8u;
typedef __attribute__((ext_vector_type(8)))  float           v8f;

// ---------- one-time fp32 -> bf16 weight conversion (full RNE) ----------

static __device__ __forceinline__ unsigned short f2bf_rne(float f) {
  unsigned int u = __float_as_uint(f);
  u += 0x7FFFu + ((u >> 16) & 1u);
  return (unsigned short)(u >> 16);
}

__global__ __launch_bounds__(256)
void f32_to_bf16_kernel(const float* __restrict__ src,
                        unsigned short* __restrict__ dst, int n4) {
  const int i = blockIdx.x * blockDim.x + threadIdx.x;
  if (i >= n4) return;
  const float4 f = reinterpret_cast<const float4*>(src)[i];
  uint2 o;
  o.x = (unsigned)f2bf_rne(f.x) | ((unsigned)f2bf_rne(f.y) << 16);
  o.y = (unsigned)f2bf_rne(f.z) | ((unsigned)f2bf_rne(f.w) << 16);
  reinterpret_cast<uint2*>(dst)[i] = o;
}

// two fp32 -> packed {bf16(hi), bf16(lo)} : one v_add each + one v_perm_b32
static __device__ __forceinline__ unsigned int pk2(float lo, float hi) {
  const unsigned int ulo = __float_as_uint(lo) + 0x7FFFu;  // round-to-nearest
  const unsigned int uhi = __float_as_uint(hi) + 0x7FFFu;
  return __builtin_amdgcn_perm(uhi, ulo, 0x07060302u);     // {uhi[3:2], ulo[3:2]}
}

// fast 1-ULP reciprocal (v_rcp_f32) -- plenty for router scores
static __device__ __forceinline__ float frcp(float x) {
  return __builtin_amdgcn_rcpf(x);
}

// bank-conflict-avoiding LDS byte offset for the bf16 x tile:
// XOR bits [7:4] of the K byte offset with the token row spreads the 16
// rows hit by lanes 0..15 across all 64 banks (16B granule preserved).
static __device__ __forceinline__ int lds_off(int row, int kbyte) {
  return row * (kHidden * 2) + (kbyte ^ ((row & 15) << 4));
}

// ---------- main fused router kernel ----------

__global__ __launch_bounds__(kThreads)
void router_bigger_kernel(const float* __restrict__ x,
                          const unsigned short* __restrict__ gate_bf,  // [256][4096] bf16
                          const unsigned short* __restrict__ up_bf,    // [256][4096] bf16
                          const float* __restrict__ extra_scale,
                          const float* __restrict__ extra_bias,
                          float* __restrict__ weights_out,
                          int*   __restrict__ indices_out) {
  __shared__ __align__(16) unsigned short s_xbf[kTokPerBlk * kHidden]; // 256 KB bf16 x tile
  __shared__ float s_prob[kTokPerBlk][kExperts + 1];                   // padded rows
  __shared__ float s_sel [kTokPerBlk][kExperts + 1];

  const int tid   = threadIdx.x;
  const int lane  = tid & 31;
  const int wave  = tid >> 5;
  const int hi    = lane >> 4;
  const int l15   = lane & 15;
  const int rtile = wave & 1;        // which 16-token tile of this block
  const int cgrp  = wave >> 1;       // 0..7, 32 columns each
  const int cb0 = cgrp * 32;
  const int cb1 = cb0 + 16;

  // ---- Phase 0: cooperative x fp32 -> bf16 staging into LDS (converted ONCE) ----
  {
    const float4* xt = reinterpret_cast<const float4*>(x + (size_t)blockIdx.x * kTokPerBlk * kHidden);
    char* xb = reinterpret_cast<char*>(s_xbf);
    #pragma unroll 8
    for (int i = 0; i < (kTokPerBlk * kHidden / 4) / kThreads; ++i) {   // 64 iters
      const int g    = tid + i * kThreads;       // float4 index, fully coalesced
      const int row  = g >> 10;                  // 1024 float4 per token row
      const int kb   = (g & 1023) * 8;           // dest byte offset along K (8B per float4)
      const float4 f = xt[g];
      uint2 o;
      o.x = pk2(f.x, f.y);
      o.y = pk2(f.z, f.w);
      *reinterpret_cast<uint2*>(xb + lds_off(row, kb)) = o;
    }
  }
  __syncthreads();

  // ---- Phase 1: WMMA K-loop; A from LDS (bf16), B from pre-converted global bf16 ----
  const char* arow = reinterpret_cast<const char*>(s_xbf);
  const int atok = rtile * 16 + l15;             // A layout: lanes 0-15 and 16-31 both M=l15
  const v16bf* bg0p = reinterpret_cast<const v16bf*>(gate_bf + (size_t)(cb0 + l15) * kHidden);
  const v16bf* bg1p = reinterpret_cast<const v16bf*>(gate_bf + (size_t)(cb1 + l15) * kHidden);
  const v16bf* bu0p = reinterpret_cast<const v16bf*>(up_bf   + (size_t)(cb0 + l15) * kHidden);
  const v16bf* bu1p = reinterpret_cast<const v16bf*>(up_bf   + (size_t)(cb1 + l15) * kHidden);

  v8f accg0 = {}, accg1 = {}, accu0 = {}, accu1 = {};

  #pragma unroll 2
  for (int k0 = 0; k0 < kHidden; k0 += 32) {
    // A fragment: lanes 0-15 (hi=0): K0..15 ; lanes 16-31 (hi=1): K16..31
    const uint4 alo = *reinterpret_cast<const uint4*>(arow + lds_off(atok, 2 * k0 + 32 * hi));
    const uint4 ahi = *reinterpret_cast<const uint4*>(arow + lds_off(atok, 2 * k0 + 16 + 32 * hi));
    v8u au;
    au[0] = alo.x; au[1] = alo.y; au[2] = alo.z; au[3] = alo.w;
    au[4] = ahi.x; au[5] = ahi.y; au[6] = ahi.z; au[7] = ahi.w;
    const v16bf a = __builtin_bit_cast(v16bf, au);

    const int bi = (k0 >> 4) + hi;               // v16bf index along K
    const v16bf bg0 = bg0p[bi];
    const v16bf bg1 = bg1p[bi];
    const v16bf bu0 = bu0p[bi];
    const v16bf bu1 = bu1p[bi];

    accg0 = __builtin_amdgcn_wmma_f32_16x16x32_bf16(false, a, false, bg0, (short)0, accg0, false, false);
    accg1 = __builtin_amdgcn_wmma_f32_16x16x32_bf16(false, a, false, bg1, (short)0, accg1, false, false);
    accu0 = __builtin_amdgcn_wmma_f32_16x16x32_bf16(false, a, false, bu0, (short)0, accu0, false, false);
    accu1 = __builtin_amdgcn_wmma_f32_16x16x32_bf16(false, a, false, bu1, (short)0, accu1, false, false);
  }

  // ---- Epilogue: SiLU(gate)*up, abs, mean over groups of 4 adjacent columns ----
  #pragma unroll
  for (int ct = 0; ct < 2; ++ct) {
    const v8f g  = ct ? accg1 : accg0;
    const v8f u  = ct ? accu1 : accu0;
    const int cb = ct ? cb1   : cb0;
    #pragma unroll
    for (int r = 0; r < 8; ++r) {
      const float gv  = g[r];
      const float sil = gv * frcp(1.0f + __expf(-gv));   // v_rcp_f32, no div expansion
      const float h   = fabsf(u[r] * sil);
      float s = h + __shfl_xor(h, 1, 32);
      s = s + __shfl_xor(s, 2, 32);
      if ((lane & 3) == 0) {
        // C/D layout: reg r, lanes 0-15 -> M=r, lanes 16-31 -> M=r+8; col = cb + (lane&15)
        s_prob[rtile * 16 + hi * 8 + r][(cb >> 2) + (l15 >> 2)] = s * 0.25f;
      }
    }
  }
  __syncthreads();

  // ---- Phase 2: per-token fp32 softmax over 64 experts + biased top-8 ----
  if (tid < kTokPerBlk) {
    const int t  = tid;
    const int gt = blockIdx.x * kTokPerBlk + t;

    float m = -3.402823466e+38f;
    for (int e = 0; e < kExperts; ++e) m = fmaxf(m, s_prob[t][e]);
    float sum = 0.0f;
    for (int e = 0; e < kExperts; ++e) {
      const float ev = __expf(s_prob[t][e] - m);
      s_prob[t][e] = ev;
      sum += ev;
    }
    const float inv = frcp(sum);
    for (int e = 0; e < kExperts; ++e) {
      const float p = s_prob[t][e] * inv;
      s_prob[t][e] = p;                       // softmax prob
      s_sel [t][e] = p + extra_bias[e];       // bias affects selection only
    }
    for (int j = 0; j < kTopK; ++j) {
      int   best = 0;
      float bv   = s_sel[t][0];
      for (int e = 1; e < kExperts; ++e) {
        const float v = s_sel[t][e];
        if (v > bv) { bv = v; best = e; }     // strict > == top_k lowest-index tie-break
      }
      s_sel[t][best] = -3.402823466e+38f;
      weights_out[gt * kTopK + j] = 1.0f + s_prob[t][best] * extra_scale[best];
      indices_out[gt * kTopK + j] = best;
    }
  }
}

extern "C" void kernel_launch(void* const* d_in, const int* in_sizes, int n_in,
                              void* d_out, int out_size, void* d_ws, size_t ws_size,
                              hipStream_t stream) {
  (void)in_sizes; (void)n_in; (void)out_size; (void)ws_size;
  const float* x           = (const float*)d_in[0];
  const float* gate_w      = (const float*)d_in[1];
  const float* up_w        = (const float*)d_in[2];
  const float* extra_scale = (const float*)d_in[3];
  const float* extra_bias  = (const float*)d_in[4];

  // workspace layout: [0, 2MB) gate bf16, [2MB, 4MB) up bf16
  unsigned short* gate_bf = (unsigned short*)d_ws;
  unsigned short* up_bf   = gate_bf + (size_t)kCols * kHidden;

  // pre-pass: convert the 8 MB of weights to bf16 once (reused by all blocks via L2)
  const int n4 = kCols * kHidden / 4;                     // 262144 float4s
  dim3 cgrid((n4 + 255) / 256), cblk(256);
  f32_to_bf16_kernel<<<cgrid, cblk, 0, stream>>>(gate_w, gate_bf, n4);
  f32_to_bf16_kernel<<<cgrid, cblk, 0, stream>>>(up_w,   up_bf,   n4);

  float* wout = (float*)d_out;                                   // [8192, 8] fp32
  int*   iout = (int*)((float*)d_out + (size_t)kTokens * kTopK); // [8192, 8] int32

  dim3 grid(kTokens / kTokPerBlk);  // 256 blocks
  dim3 block(kThreads);             // 512 threads = 16 waves
  router_bigger_kernel<<<grid, block, 0, stream>>>(x, gate_bf, up_bf,
                                                   extra_scale, extra_bias,
                                                   wout, iout);
}
